// PLAggregator_33878702031556
// MI455X (gfx1250) — matrix operand order repaired and checked by reading, not compiled, not run on hardware
//
#include <hip/hip_runtime.h>
#include <hip/hip_bf16.h>

// ---------------------------------------------------------------------------
// GraphRec aggregator for MI455X (gfx1250, wave32, WMMA).
// One workgroup per batch row b: 64x128 token tile (H=50 padded to 64),
// 4 back-to-back bf16 WMMA GEMMs + fused logit projection + softmax + pooling.
// ---------------------------------------------------------------------------

typedef __attribute__((ext_vector_type(16))) __bf16 v16bf;
typedef __attribute__((ext_vector_type(8)))  __bf16 v8bf;
typedef __attribute__((ext_vector_type(8)))  float  v8f;

#define DMODEL 128
#define XA_STRIDE 264   // 256 bf16 + 8 pad (528B row: 16B aligned, 4-dword bank skew)
#define XB_STRIDE 136   // 128 bf16 + 8 pad (272B row)

__device__ __forceinline__ unsigned short f2bf(float f) {
    __bf16 h = (__bf16)f;
    return __builtin_bit_cast(unsigned short, h);
}
__device__ __forceinline__ float bf2f(unsigned short u) {
    return (float)__builtin_bit_cast(__bf16, u);
}

// ---------------------------------------------------------------------------
// Weight preprocessing: fp32 KxN row-major -> bf16 NxK row-major (transposed).
// Tiny (<=32K elems per matrix), runs once per launch, output is L2-resident
// and shared by all 4096 workgroups of the main kernel.
// ---------------------------------------------------------------------------
__global__ void wtrans_kernel(const float* __restrict__ in,
                              unsigned short* __restrict__ out, int K, int N) {
    int i = blockIdx.x * blockDim.x + threadIdx.x;
    if (i >= K * N) return;
    int n = i / K;
    int k = i - n * K;
    out[i] = f2bf(in[k * N + n]);   // out[n*K+k] = in[k*N+n]
}

// ---------------------------------------------------------------------------
// Per-wave GEMM: C[64x16] = A[64xK] (LDS bf16, stride As) x Wt^T tile.
// Wt is NxK bf16 row-major in global. Wave covers column tile nb..nb+15.
// A-fragment K order used here (contiguous; B uses the same): lanes 0-15 row m, K {0..15};
// lanes 16-31 row m, K {16..31}  -> two ds_load_b128 per fragment.
// B-fragment: lane n = nb+(l&15), 16 contiguous K (+16 for upper half)
//   -> 32 contiguous bytes from Wt -> two global_load_b128.
// ---------------------------------------------------------------------------
template <int K>
__device__ __forceinline__ void gemm_tile(const unsigned short* __restrict__ A, int As,
                                          const unsigned short* __restrict__ Wt,
                                          int nb, int lane, v8f acc[4]) {
#pragma unroll
    for (int r = 0; r < 4; ++r) acc[r] = v8f{0.f, 0.f, 0.f, 0.f, 0.f, 0.f, 0.f, 0.f};

    const int n = nb + (lane & 15);
    const unsigned short* brow = Wt + n * K + ((lane >> 4) << 4);   // +16 K for hi half
    const unsigned short* arow = A + (lane & 15) * As + ((lane >> 4) << 4); // +16 K hi half

#pragma unroll
    for (int k = 0; k < K; k += 32) {
        v16bf bf = *(const v16bf*)(brow + k);        // 32B contiguous -> 2x b128
#pragma unroll
        for (int r = 0; r < 4; ++r) {
            const unsigned short* p = arow + r * 16 * As + k;
            v8bf lo = *(const v8bf*)(p);             // ds_load_b128
            v8bf hi = *(const v8bf*)(p + 8);         // ds_load_b128
            v16bf af = __builtin_shufflevector(lo, hi, 0, 1, 2, 3, 4, 5, 6, 7,
                                               8, 9, 10, 11, 12, 13, 14, 15);
            acc[r] = __builtin_amdgcn_wmma_f32_16x16x32_bf16(
                false, af, false, bf, (short)0, acc[r], false, false);
        }
    }
}

// Accumulator layout: lane l holds column nb+(l&15); VGPR j holds row
// r*16 + ((l>=16)?8:0) + j. Apply bias + ReLU, store bf16 to LDS row-major.
__device__ __forceinline__ void store_relu(const v8f acc[4], const float* __restrict__ bias,
                                           unsigned short* __restrict__ dst, int Ds,
                                           int nb, int lane) {
    const int col = nb + (lane & 15);
    const float bv = bias[col];
    const int ro = (lane >> 4) << 3;
#pragma unroll
    for (int r = 0; r < 4; ++r)
#pragma unroll
        for (int j = 0; j < 8; ++j) {
            float v = fmaxf(acc[r][j] + bv, 0.f);
            dst[(r * 16 + ro + j) * Ds + col] = f2bf(v);
        }
}

// ---------------------------------------------------------------------------
// Main kernel: grid = B blocks, 256 threads (8 wave32).
// ---------------------------------------------------------------------------
__global__ __launch_bounds__(256) void graphrec_kernel(
    const int* __restrict__ nodes_u, const int* __restrict__ hist_ids,
    const int* __restrict__ hist_labels,
    const float* __restrict__ v2e, const float* __restrict__ u2e,
    const float* __restrict__ r2e,
    const unsigned short* __restrict__ W1t, const float* __restrict__ b1,
    const unsigned short* __restrict__ W2t, const float* __restrict__ b2,
    const unsigned short* __restrict__ A1t, const float* __restrict__ ab1,
    const unsigned short* __restrict__ A2t, const float* __restrict__ ab2,
    const float* __restrict__ att3, int H, float* __restrict__ out) {

    __shared__ unsigned short sXA[64 * XA_STRIDE];  // [e_uv|e_r] -> later [o|u]
    __shared__ unsigned short sXB[64 * XB_STRIDE];  // intermediate activations
    __shared__ float sU[DMODEL];                    // user embedding fp32
    __shared__ float sL[64];                        // logits -> softmax weights

    const int b    = blockIdx.x;
    const int tid  = threadIdx.x;
    const int lane = tid & 31;
    const int wv   = tid >> 5;
    const int nb   = wv * 16;   // this wave's output column tile

    // ---- Phase 0: gathers (fp32 global -> bf16 LDS) -----------------------
    const int uid = nodes_u[b];
    if (tid < DMODEL) sU[tid] = u2e[(size_t)uid * DMODEL + tid];
    if (tid < 64) sL[tid] = 0.f;

    for (int r = wv; r < 64; r += 8) {              // r is wave-uniform
        float4 ev = {0.f, 0.f, 0.f, 0.f};
        float4 er = {0.f, 0.f, 0.f, 0.f};
        if (r < H) {
            int iid = hist_ids[(size_t)b * H + r];     // scalarized (uniform)
            int rid = hist_labels[(size_t)b * H + r];
            ev = *(const float4*)(v2e + (size_t)iid * DMODEL + lane * 4);
            er = *(const float4*)(r2e + (size_t)rid * DMODEL + lane * 4);
        }
        unsigned short* d0 = &sXA[r * XA_STRIDE + lane * 4];
        d0[0] = f2bf(ev.x); d0[1] = f2bf(ev.y); d0[2] = f2bf(ev.z); d0[3] = f2bf(ev.w);
        unsigned short* d1 = &sXA[r * XA_STRIDE + DMODEL + lane * 4];
        d1[0] = f2bf(er.x); d1[1] = f2bf(er.y); d1[2] = f2bf(er.z); d1[3] = f2bf(er.w);
    }
    __syncthreads();

    v8f acc[4];

    // ---- GEMM1: relu([e_uv|e_r] @ W1 + b1) -> sXB -------------------------
    gemm_tile<2 * DMODEL>(sXA, XA_STRIDE, W1t, nb, lane, acc);
    store_relu(acc, b1, sXB, XB_STRIDE, nb, lane);
    __syncthreads();

    // broadcast user embedding into sXA[:,128:256] (e_uv/e_r now dead)
    for (int i = tid; i < 64 * DMODEL; i += 256) {
        int r = i >> 7, d = i & (DMODEL - 1);
        sXA[r * XA_STRIDE + DMODEL + d] = f2bf(sU[d]);
    }

    // ---- GEMM2: o = relu(x1 @ W2 + b2) -> sXA[:,0:128] --------------------
    gemm_tile<DMODEL>(sXB, XB_STRIDE, W2t, nb, lane, acc);
    store_relu(acc, b2, sXA, XA_STRIDE, nb, lane);
    __syncthreads();

    // ---- GEMM3: relu([o|u] @ A1 + ab1) -> sXB -----------------------------
    gemm_tile<2 * DMODEL>(sXA, XA_STRIDE, A1t, nb, lane, acc);
    store_relu(acc, ab1, sXB, XB_STRIDE, nb, lane);
    __syncthreads();

    // ---- GEMM4 + fused 128->1 projection: logits into sL ------------------
    gemm_tile<DMODEL>(sXB, XB_STRIDE, A2t, nb, lane, acc);
    {
        const int col = nb + (lane & 15);
        const float bv = ab2[col];
        const float a3 = att3[col];
        const int ro = (lane >> 4) << 3;
#pragma unroll
        for (int r = 0; r < 4; ++r)
#pragma unroll
            for (int j = 0; j < 8; ++j) {
                float v = fmaxf(acc[r][j] + bv, 0.f);
                atomicAdd(&sL[r * 16 + ro + j], v * a3);   // ds_add_f32
            }
    }
    __syncthreads();

    // ---- softmax over H logits (wave 0; shift-invariant so att3_b skipped) -
    if (wv == 0) {
        float v0 = (lane < H) ? sL[lane] : -3.4e38f;
        float v1 = (lane + 32 < H) ? sL[lane + 32] : -3.4e38f;
        float m = fmaxf(v0, v1);
#pragma unroll
        for (int off = 16; off > 0; off >>= 1) m = fmaxf(m, __shfl_xor(m, off));
        float e0 = (lane < H) ? __expf(v0 - m) : 0.f;
        float e1 = (lane + 32 < H) ? __expf(v1 - m) : 0.f;
        float s = e0 + e1;
#pragma unroll
        for (int off = 16; off > 0; off >>= 1) s += __shfl_xor(s, off);
        float inv = 1.f / s;
        sL[lane] = e0 * inv;
        sL[lane + 32] = e1 * inv;
    }
    __syncthreads();

    // ---- attention-weighted pooling + average with user embedding ---------
    if (tid < DMODEL) {
        int d = tid;
        float s = 0.f;
        for (int h = 0; h < H; ++h)
            s += sL[h] * bf2f(sXA[h * XA_STRIDE + d]);
        out[(size_t)b * DMODEL + d] = 0.5f * (s + sU[d]);
    }
}

// ---------------------------------------------------------------------------
// Host launcher
// ---------------------------------------------------------------------------
extern "C" void kernel_launch(void* const* d_in, const int* in_sizes, int n_in,
                              void* d_out, int out_size, void* d_ws, size_t ws_size,
                              hipStream_t stream) {
    const int*   nodes_u     = (const int*)d_in[0];
    const int*   hist_ids    = (const int*)d_in[1];
    const int*   hist_labels = (const int*)d_in[2];
    const float* v2e   = (const float*)d_in[3];
    const float* u2e   = (const float*)d_in[4];
    const float* r2e   = (const float*)d_in[5];
    const float* w1    = (const float*)d_in[6];
    const float* b1    = (const float*)d_in[7];
    const float* w2    = (const float*)d_in[8];
    const float* b2    = (const float*)d_in[9];
    const float* a1    = (const float*)d_in[10];
    const float* ab1   = (const float*)d_in[11];
    const float* a2    = (const float*)d_in[12];
    const float* ab2   = (const float*)d_in[13];
    const float* att3  = (const float*)d_in[14];
    // d_in[15] = att3_b: constant shift of all logits, softmax-invariant.

    const int B = in_sizes[0];
    const int H = in_sizes[1] / B;           // 50

    // Workspace: transposed bf16 weights (N x K row-major).
    unsigned short* W1t = (unsigned short*)d_ws;          // 128 x 256
    unsigned short* W2t = W1t + 128 * 256;                // 128 x 128
    unsigned short* A1t = W2t + 128 * 128;                // 128 x 256
    unsigned short* A2t = A1t + 128 * 256;                // 128 x 128

    const int thr = 256;
    wtrans_kernel<<<(256 * 128 + thr - 1) / thr, thr, 0, stream>>>(w1, W1t, 256, 128);
    wtrans_kernel<<<(128 * 128 + thr - 1) / thr, thr, 0, stream>>>(w2, W2t, 128, 128);
    wtrans_kernel<<<(256 * 128 + thr - 1) / thr, thr, 0, stream>>>(a1, A1t, 256, 128);
    wtrans_kernel<<<(128 * 128 + thr - 1) / thr, thr, 0, stream>>>(a2, A2t, 128, 128);

    graphrec_kernel<<<B, 256, 0, stream>>>(
        nodes_u, hist_ids, hist_labels, v2e, u2e, r2e,
        W1t, b1, W2t, b2, A1t, ab1, A2t, ab2, att3, H, (float*)d_out);
}
